// RelationGraphConv_56573309223578
// MI455X (gfx1250) — matrix operand order repaired and checked by reading, not compiled, not run on hardware
//
#include <hip/hip_runtime.h>
#include <hip/hip_bf16.h>

typedef __attribute__((ext_vector_type(16))) _Float16 v16h;
typedef __attribute__((ext_vector_type(8)))  _Float16 v8h;
typedef __attribute__((ext_vector_type(8)))  float    v8f;

#define DFEAT 256

// ---------------------------------------------------------------------------
// Elementwise helpers
// ---------------------------------------------------------------------------
__global__ void k_zero_f32(float* __restrict__ p, int n) {
    int i = blockIdx.x * blockDim.x + threadIdx.x;
    int stride = gridDim.x * blockDim.x;
    for (; i < n; i += stride) p[i] = 0.0f;
}

__global__ void k_cvt_f32_to_f16(const float* __restrict__ src,
                                 _Float16* __restrict__ dst, int n) {
    int i = blockIdx.x * blockDim.x + threadIdx.x;
    int stride = gridDim.x * blockDim.x;
    for (; i < n; i += stride) dst[i] = (_Float16)src[i];
}

// W [R][I][O] f32  ->  Wt [R][O][I] f16   (transpose inner 256x256)
__global__ void k_cvt_transpose_w(const float* __restrict__ w,
                                  _Float16* __restrict__ wt, int n) {
    int i = blockIdx.x * blockDim.x + threadIdx.x;
    int stride = gridDim.x * blockDim.x;
    for (; i < n; i += stride) {
        int r   = i >> 16;          // /65536
        int rem = i & 0xFFFF;
        int o   = rem >> 8;         // output channel
        int in  = rem & 0xFF;       // input channel
        wt[i] = (_Float16)w[(r << 16) + (in << 8) + o];  // wt[r][o][in]
    }
}

__global__ void k_bias_relu(float* __restrict__ out,
                            const float* __restrict__ bias, int n) {
    int i = blockIdx.x * blockDim.x + threadIdx.x;
    int stride = gridDim.x * blockDim.x;
    for (; i < n; i += stride) {
        float v = out[i] + bias[i & (DFEAT - 1)];
        out[i] = v > 0.0f ? v : 0.0f;
    }
}

// ---------------------------------------------------------------------------
// Per-relation dense transform:  xt[n, 256] = x_f16[n, 256] @ Wt_r^T
// Wt_r is [O=256][I=256] f16 (pre-transposed), so a B fragment is one
// contiguous 32-byte load.  Each wave computes a 32x64 output strip
// (2 M-tiles x 4 N-tiles): every B fragment feeds two WMMAs, cutting the
// load:WMMA ratio to 1.5 B128 loads per V_WMMA_F32_16X16X32_F16.
// ---------------------------------------------------------------------------
__global__ void __launch_bounds__(256)
k_gemm_f16(const _Float16* __restrict__ xh,   // [n_nodes][256]
           const _Float16* __restrict__ wt,   // [256][256] this relation, [O][I]
           float* __restrict__ xt,            // [n_nodes][256]
           int n_nodes) {
    const int lane = threadIdx.x & 31;
    const int wave = threadIdx.x >> 5;
    const int m0   = (blockIdx.x * 8 + wave) * 32;   // 32 rows per wave
    if (m0 >= n_nodes) return;                        // wave-uniform exit
    const int grp = lane >> 4;                        // 0 | 1
    const int ln  = lane & 15;
    const int n0  = blockIdx.y * 64;                  // 64-wide N strip

    // A rows for the two M-tiles (clamped so EXEC stays all-1s for WMMA)
    int arow0 = m0 + ln;      if (arow0 >= n_nodes) arow0 = n_nodes - 1;
    int arow1 = m0 + 16 + ln; if (arow1 >= n_nodes) arow1 = n_nodes - 1;
    const _Float16* ap0 = xh + (size_t)arow0 * DFEAT;
    const _Float16* ap1 = xh + (size_t)arow1 * DFEAT;

    v8f acc00 = {}, acc01 = {}, acc02 = {}, acc03 = {};
    v8f acc10 = {}, acc11 = {}, acc12 = {}, acc13 = {};

    #pragma unroll
    for (int k0 = 0; k0 < DFEAT; k0 += 32) {
        // A fragments: halves [k0+16g .. +7] and [k0+16g+8 .. +7] of each row
        v8h a0lo = *(const v8h*)(ap0 + k0 + 16 * grp);
        v8h a0hi = *(const v8h*)(ap0 + k0 + 16 * grp + 8);
        v8h a1lo = *(const v8h*)(ap1 + k0 + 16 * grp);
        v8h a1hi = *(const v8h*)(ap1 + k0 + 16 * grp + 8);
        v16h a0, a1;
        #pragma unroll
        for (int i = 0; i < 8; ++i) {
            a0[i] = a0lo[i]; a0[i + 8] = a0hi[i];
            a1[i] = a1lo[i]; a1[i + 8] = a1hi[i];
        }

        // B fragments: 16 consecutive halves of Wt row (n0+t*16+ln), K off k0+16g
        const _Float16* bbase = wt + k0 + 16 * grp;
        v16h b0 = *(const v16h*)(bbase + (size_t)(n0 + 0  + ln) * DFEAT);
        v16h b1 = *(const v16h*)(bbase + (size_t)(n0 + 16 + ln) * DFEAT);
        v16h b2 = *(const v16h*)(bbase + (size_t)(n0 + 32 + ln) * DFEAT);
        v16h b3 = *(const v16h*)(bbase + (size_t)(n0 + 48 + ln) * DFEAT);

        acc00 = __builtin_amdgcn_wmma_f32_16x16x32_f16(false, a0, false, b0, (short)0, acc00, false, false);
        acc10 = __builtin_amdgcn_wmma_f32_16x16x32_f16(false, a1, false, b0, (short)0, acc10, false, false);
        acc01 = __builtin_amdgcn_wmma_f32_16x16x32_f16(false, a0, false, b1, (short)0, acc01, false, false);
        acc11 = __builtin_amdgcn_wmma_f32_16x16x32_f16(false, a1, false, b1, (short)0, acc11, false, false);
        acc02 = __builtin_amdgcn_wmma_f32_16x16x32_f16(false, a0, false, b2, (short)0, acc02, false, false);
        acc12 = __builtin_amdgcn_wmma_f32_16x16x32_f16(false, a1, false, b2, (short)0, acc12, false, false);
        acc03 = __builtin_amdgcn_wmma_f32_16x16x32_f16(false, a0, false, b3, (short)0, acc03, false, false);
        acc13 = __builtin_amdgcn_wmma_f32_16x16x32_f16(false, a1, false, b3, (short)0, acc13, false, false);
    }

    // Store: element e of v8f -> row mBase+e+8g, col n0+t*16+ln
    #pragma unroll
    for (int e = 0; e < 8; ++e) {
        int row0 = m0 + e + 8 * grp;
        if (row0 < n_nodes) {
            float* orow = xt + (size_t)row0 * DFEAT + n0;
            orow[0  + ln] = acc00[e];
            orow[16 + ln] = acc01[e];
            orow[32 + ln] = acc02[e];
            orow[48 + ln] = acc03[e];
        }
        int row1 = m0 + 16 + e + 8 * grp;
        if (row1 < n_nodes) {
            float* orow = xt + (size_t)row1 * DFEAT + n0;
            orow[0  + ln] = acc10[e];
            orow[16 + ln] = acc11[e];
            orow[32 + ln] = acc12[e];
            orow[48 + ln] = acc13[e];
        }
    }
}

// ---------------------------------------------------------------------------
// Edge scatter for one relation: out[row] += xt[col]   (xt is L2-resident)
// One wave per edge; 8 f32 atomics per lane (256 per edge).
// ---------------------------------------------------------------------------
__global__ void __launch_bounds__(256)
k_scatter(const int* __restrict__ rows, const int* __restrict__ cols,
          const int* __restrict__ etype, const float* __restrict__ xt,
          float* __restrict__ out, int rel, int n_edges) {
    const int lane   = threadIdx.x & 31;
    const int wave   = blockIdx.x * (blockDim.x >> 5) + (threadIdx.x >> 5);
    const int nwaves = gridDim.x * (blockDim.x >> 5);
    for (int e = wave; e < n_edges; e += nwaves) {
        if (etype[e] != rel) continue;            // wave-uniform branch
        const int row = rows[e];
        const int col = cols[e];
        const float4* src = (const float4*)(xt + (size_t)col * DFEAT);
        float*        dst = out + (size_t)row * DFEAT;
        float4 v0 = src[lane];
        float4 v1 = src[lane + 32];
        int c0 = lane * 4, c1 = 128 + lane * 4;
        atomicAdd(dst + c0 + 0, v0.x);
        atomicAdd(dst + c0 + 1, v0.y);
        atomicAdd(dst + c0 + 2, v0.z);
        atomicAdd(dst + c0 + 3, v0.w);
        atomicAdd(dst + c1 + 0, v1.x);
        atomicAdd(dst + c1 + 1, v1.y);
        atomicAdd(dst + c1 + 2, v1.z);
        atomicAdd(dst + c1 + 3, v1.w);
    }
}

// ---------------------------------------------------------------------------
// Host-side launcher
// ---------------------------------------------------------------------------
extern "C" void kernel_launch(void* const* d_in, const int* in_sizes, int n_in,
                              void* d_out, int out_size, void* d_ws, size_t ws_size,
                              hipStream_t stream) {
    const float* d_x     = (const float*)d_in[0];   // [N,256]
    const int*   d_ei    = (const int*)d_in[1];     // [2,E] flat
    const int*   d_et    = (const int*)d_in[2];     // [E]
    const float* d_w     = (const float*)d_in[3];   // [8,256,256]
    const float* d_bias  = (const float*)d_in[4];   // [256]
    float*       out     = (float*)d_out;

    const int n_nodes = in_sizes[0] / DFEAT;        // 50000
    const int n_edges = in_sizes[2];                // 800000
    const int n_rel   = in_sizes[3] / (DFEAT * DFEAT); // 8
    const int nx      = n_nodes * DFEAT;            // 12.8M
    const int nw      = n_rel * DFEAT * DFEAT;      // 524288

    // workspace layout
    char* ws = (char*)d_ws;
    _Float16* xh = (_Float16*)ws;                                   // 25.6 MB
    _Float16* wt = (_Float16*)(ws + (size_t)nx * sizeof(_Float16)); // 1 MB
    float*    xt = (float*)(ws + (size_t)nx * sizeof(_Float16)
                               + (size_t)nw * sizeof(_Float16));    // 51.2 MB

    const int* rows = d_ei;
    const int* cols = d_ei + n_edges;

    // 1) zero the accumulation target (d_out is poisoned by the harness)
    k_zero_f32<<<2048, 256, 0, stream>>>(out, nx);
    // 2) convert x -> f16
    k_cvt_f32_to_f16<<<2048, 256, 0, stream>>>(d_x, xh, nx);
    // 3) convert + transpose W -> f16 [R][O][I]
    k_cvt_transpose_w<<<1024, 256, 0, stream>>>(d_w, wt, nw);

    // 4) per relation: dense WMMA transform into L2-resident xt, then scatter
    dim3 ggrid((n_nodes + 255) / 256, DFEAT / 64);  // (196, 4); 8 waves/block
    for (int r = 0; r < n_rel; ++r) {
        k_gemm_f16<<<ggrid, 256, 0, stream>>>(
            xh, wt + (size_t)r * DFEAT * DFEAT, xt, n_nodes);
        k_scatter<<<1024, 256, 0, stream>>>(
            rows, cols, d_et, xt, out, r, n_edges);
    }

    // 5) bias + ReLU in place
    k_bias_relu<<<2048, 256, 0, stream>>>(out, d_bias, nx);
}
